// HomoAttention_62629213110490
// MI455X (gfx1250) — matrix operand order repaired and checked by reading, not compiled, not run on hardware
//
#include <hip/hip_runtime.h>
#include <hip/hip_bf16.h>

// HomoAttention: multi-facet cosine attention.
//   features      [128,16,256]      f32   (B=128, N=16, F=8 facets, E=32)
//   neigh_features[128,16,64,256]   f32   (K=64 neighbors)
//   out           [128,16,256]      f32
//
// Memory-bound: 134 MB of z streamed once (~6 us at 23.3 TB/s on MI455X).
// One workgroup per (b,n): stage z (64KB, padded) in LDS via async-to-LDS
// loads (ASYNCcnt) when available, then do both einsums as
// V_WMMA_F32_16X16X32_F16 with one broadcast operand (f16 in / f32 accum).

typedef __attribute__((ext_vector_type(16))) _Float16 v16h;
typedef __attribute__((ext_vector_type(8)))  float    v8f;
typedef __attribute__((ext_vector_type(4)))  int      v4i;

#define F_  8
#define E_  32
#define K_  64
#define D_  256
#define ZST 260   // padded LDS row stride in floats (kills stride-256 bank aliasing)

#if defined(__gfx1250__) && __has_builtin(__builtin_amdgcn_global_load_async_to_lds_b128) && __has_builtin(__builtin_amdgcn_s_wait_asynccnt)
#define USE_ASYNC_LDS 1
typedef __attribute__((address_space(1))) v4i GV4;   // global b128 chunk
typedef __attribute__((address_space(3))) v4i LV4;   // LDS b128 chunk
#else
#define USE_ASYNC_LDS 0
#endif

__device__ __forceinline__ float wave_max32(float v) {
#pragma unroll
  for (int m = 16; m >= 1; m >>= 1) v = fmaxf(v, __shfl_xor(v, m, 32));
  return v;
}
__device__ __forceinline__ float wave_sum32(float v) {
#pragma unroll
  for (int m = 16; m >= 1; m >>= 1) v += __shfl_xor(v, m, 32);
  return v;
}

// A-matrix (16-bit, 16x32) k-dim index for half-pair j (0..7), lane-half hi:
//   VGPR0..3: K = {0,1},{2,3},{4,5},{6,7}  (lanes0-15)  +8 for lanes16-31
//   VGPR4..7: K = 16..23 (lanes0-15) | 24..31 (lanes16-31)
__device__ __forceinline__ int amap(int j, int hi) {
  return ((j & 4) << 2) + (hi << 3) + ((j & 3) << 1);
}
// B-matrix (16-bit, 32x16) row index for half-pair j, lane-half hi
// (same K order as amap: VGPR0..3 K=0..7 | 8..15, VGPR4..7 K=16..23 | 24..31).
// Used only with broadcast columns here.
__device__ __forceinline__ int bmap(int j, int hi) {
  return ((j & 4) << 2) + (hi << 3) + ((j & 3) << 1);
}

__global__ __launch_bounds__(256)
void homo_attention_kernel(const float* __restrict__ xg_all,
                           const float* __restrict__ zg_all,
                           float* __restrict__ out) {
  const int bn   = blockIdx.x;          // 0..2047  = b*16+n
  const int t    = threadIdx.x;         // 0..255
  const int lane = t & 31;
  const int wave = t >> 5;              // 8 waves -> 8 facets
  const int m    = lane & 15;           // row-in-tile
  const int hi   = lane >> 4;           // lane half

  const float* xg = xg_all + (size_t)bn * D_;
  const float* zg = zg_all + (size_t)bn * K_ * D_;

  __shared__ __align__(16) float zs[K_ * ZST];   // raw z, padded rows  (~65 KB)
  __shared__ float xs[D_];                       // normalized x  [f*32+e]
  __shared__ float invs[K_ * F_];                // 1/||z||       [k*8+f]
  __shared__ float sS[F_][K_];                   // scores
  __shared__ float wbuf[F_][K_];                 // softmax(a)*inv
  __shared__ float ubuf[D_];                     // output staging [f*32+e]

  // ---- Phase 2a: z -> LDS.  Prefer GLOBAL_LOAD_ASYNC_TO_LDS_B128 (no VGPR
  //      round-trip, tracked by ASYNCcnt); fallback: float4 copy. ----
  {
    const float4* zg4 = (const float4*)zg;
    float4*       zs4 = (float4*)zs;             // ZST=260 is a multiple of 4
#if USE_ASYNC_LDS
#pragma unroll
    for (int i = 0; i < 16; ++i) {
      int idx = i * 256 + t;                     // 4096 float4s total
      int k   = idx >> 6;                        // 64 float4 per 256-float row
      int c   = idx & 63;
      GV4* gsrc = (GV4*)(zg4 + idx);
      LV4* ldst = (LV4*)(zs4 + k * (ZST / 4) + c);
      __builtin_amdgcn_global_load_async_to_lds_b128(gsrc, ldst, 0, 0);
    }
#else
#pragma unroll
    for (int i = 0; i < 16; ++i) {
      int idx = i * 256 + t;
      int k   = idx >> 6;
      int c   = idx & 63;
      zs4[k * (ZST / 4) + c] = zg4[idx];
    }
#endif
  }

  // ---- Phase 1 (overlaps async copy): normalize x (wave w = facet w) ----
  {
    float v  = xg[t];
    float ss = wave_sum32(v * v);
    float inv = 1.0f / fmaxf(sqrtf(ss), 1e-12f);
    xs[t] = v * inv;
  }

#if USE_ASYNC_LDS
  __builtin_amdgcn_s_wait_asynccnt(0);           // this wave's LDS writes landed
#endif
  __syncthreads();

  // ---- Phase 2b: per-(k,f) inverse norms (512 vectors, 2 per thread) ----
#pragma unroll
  for (int r = 0; r < 2; ++r) {
    int vid = t + r * 256;                       // = k*8+f
    int k = vid >> 3, f = vid & 7;
    const float4* p = (const float4*)&zs[k * ZST + f * E_];
    float ss = 0.f;
#pragma unroll
    for (int q = 0; q < 8; ++q) {
      float4 v = p[q];
      ss += v.x * v.x + v.y * v.y + v.z * v.z + v.w * v.w;
    }
    invs[vid] = 1.0f / fmaxf(sqrtf(ss), 1e-12f);
  }
  __syncthreads();

  const int f = wave;                            // this wave's facet

  // ---- Phase 3: scores via WMMA.  A = z_raw[16k x 32e], B = x_hat bcast ----
  v16h bx;
#pragma unroll
  for (int j = 0; j < 8; ++j) {
    int e = bmap(j, hi);
    bx[2 * j]     = (_Float16)xs[f * E_ + e];
    bx[2 * j + 1] = (_Float16)xs[f * E_ + e + 1];
  }
#pragma unroll
  for (int tile = 0; tile < 4; ++tile) {
    int k0 = tile * 16;
    v16h a;
#pragma unroll
    for (int j = 0; j < 8; ++j) {
      int e = amap(j, hi);
      const float* row = &zs[(k0 + m) * ZST + f * E_ + e];
      a[2 * j]     = (_Float16)row[0];
      a[2 * j + 1] = (_Float16)row[1];
    }
    v8f c = {};
    c = __builtin_amdgcn_wmma_f32_16x16x32_f16(false, a, false, bx,
                                               (short)0, c, false, false);
    if (m == 0) {                                // lanes 0 & 16 hold rows r(+8)
#pragma unroll
      for (int r = 0; r < 8; ++r) {
        int k = k0 + r + (hi << 3);
        sS[f][k] = c[r] * invs[k * F_ + f];      // fold in z normalization
      }
    }
  }
  __syncthreads();

  // ---- Phase 4: softmax over 64 neighbors (per wave), fold in inv ----
  {
    float s0 = sS[f][lane], s1 = sS[f][lane + 32];
    float mx = wave_max32(fmaxf(s0, s1));
    float e0 = __expf(s0 - mx), e1 = __expf(s1 - mx);
    float dn = wave_sum32(e0 + e1);
    float rdn = 1.0f / dn;
    wbuf[f][lane]      = e0 * rdn * invs[lane * F_ + f];
    wbuf[f][lane + 32] = e1 * rdn * invs[(lane + 32) * F_ + f];
  }
  __syncthreads();

  // ---- Phase 5: u = w^T Z via WMMA.  A = z^T[16e x 32k], B = w bcast ----
#pragma unroll
  for (int et = 0; et < 2; ++et) {
    v8f c = {};
#pragma unroll
    for (int kc = 0; kc < 2; ++kc) {
      v16h a, b;
#pragma unroll
      for (int j = 0; j < 8; ++j) {
        int kk = kc * 32 + amap(j, hi);
        a[2 * j]     = (_Float16)zs[kk * ZST + f * E_ + et * 16 + m];
        a[2 * j + 1] = (_Float16)zs[(kk + 1) * ZST + f * E_ + et * 16 + m];
        int kb = kc * 32 + bmap(j, hi);
        b[2 * j]     = (_Float16)wbuf[f][kb];
        b[2 * j + 1] = (_Float16)wbuf[f][kb + 1];
      }
      c = __builtin_amdgcn_wmma_f32_16x16x32_f16(false, a, false, b,
                                                 (short)0, c, false, false);
    }
    if (m == 0) {
#pragma unroll
      for (int r = 0; r < 8; ++r)
        ubuf[f * E_ + et * 16 + r + (hi << 3)] = c[r];
    }
  }
  __syncthreads();

  // ---- Phase 6: coalesced store ----
  out[(size_t)bn * D_ + t] = ubuf[t];
}

extern "C" void kernel_launch(void* const* d_in, const int* in_sizes, int n_in,
                              void* d_out, int out_size, void* d_ws, size_t ws_size,
                              hipStream_t stream) {
  const float* features = (const float*)d_in[0];   // [128,16,256]
  const float* neigh    = (const float*)d_in[1];   // [128,16,64,256]
  float* out            = (float*)d_out;           // [128,16,256]
  int n_bn = in_sizes[0] / D_;                     // 2048 (b,n) pairs
  homo_attention_kernel<<<n_bn, 256, 0, stream>>>(features, neigh, out);
}
